// Layer_82738249990805
// MI455X (gfx1250) — matrix operand order repaired and checked by reading, not compiled, not run on hardware
//
#include <hip/hip_runtime.h>
#include <hip/hip_bf16.h>

// ---------------- problem constants ----------------
constexpr int kB  = 4;
constexpr int kS  = 1024;
constexpr int kH  = 768;
constexpr int kNH = 12;
constexpr int kHD = 64;
constexpr int kFF = 3072;
constexpr int kM  = kB * kS;          // 4096 rows

typedef __bf16 bf16;
typedef __attribute__((ext_vector_type(16))) __bf16 v16bf;
typedef __attribute__((ext_vector_type(8)))  __bf16 v8bf;
typedef __attribute__((ext_vector_type(8)))  float  v8f;
typedef __attribute__((ext_vector_type(4)))  unsigned int v4u;
typedef __attribute__((ext_vector_type(8)))  int v8i;
typedef __attribute__((ext_vector_type(4)))  int v4i;
typedef unsigned int u32;

union V16U { v16bf v; v8bf h[2]; };

// ---- WMMA fragment loads from LDS (wave32, contiguous k = 16*hi + s in both A and B) ----
__device__ __forceinline__ v16bf frag_a_lds(const bf16* base, int lane, int ls) {
  const int m = lane & 15, hi = lane >> 4;
  const bf16* row = base + m * ls;
  V16U u;
  u.h[0] = *(const v8bf*)(row + hi * 16);       // K = hi*16 .. +7
  u.h[1] = *(const v8bf*)(row + hi * 16 + 8);   // K = hi*16+8 .. +15
  return u.v;
}
__device__ __forceinline__ v16bf frag_b_lds(const bf16* base, int lane, int ls) {
  const int n = lane & 15, hi = lane >> 4;
  const bf16* col = base + n * ls + hi * 16;    // column-major [n][k]
  V16U u;
  u.h[0] = *(const v8bf*)(col);
  u.h[1] = *(const v8bf*)(col + 8);
  return u.v;
}
__device__ __forceinline__ v8f wmma_bf16(v16bf a, v16bf b, v8f c) {
  return __builtin_amdgcn_wmma_f32_16x16x32_bf16(false, a, false, b,
                                                 (short)0, c, false, false);
}

// ---- CDNA5 async global->LDS copy (ASYNCcnt-tracked, no VGPR round trip) ---
__device__ __forceinline__ void async_copy_b128(const void* gptr, void* lptr) {
  // flat LDS aperture: addr[31:0] is the LDS byte offset (ISA 10.2)
  uint32_t lds_off = (uint32_t)(uintptr_t)lptr;
  uint64_t gaddr   = (uint64_t)(uintptr_t)gptr;
  asm volatile("global_load_async_to_lds_b128 %0, %1, off"
               :: "v"(lds_off), "v"(gaddr) : "memory");
}
__device__ __forceinline__ void wait_async() {
  asm volatile("s_wait_asynccnt 0" ::: "memory");
}

// ---- CDNA5 Tensor Data Mover: 2D bf16 tile -> LDS with row padding --------
// Loads tile_y rows of tile_x bf16 elements (row stride = row_stride elems in
// global memory) into LDS row-major with 8-element padding per row (LDS row
// stride = 72 bf16): pad_interval code 4 = 32 DWORDs, pad_amount code 3 = 4
// DWORDs (ISA 8.4). D# layout per ISA 8.3/8.4; TENSORcnt-tracked; EXEC ignored.
__device__ __forceinline__ void tdm_load_2d(const void* gptr, void* lptr,
                                            u32 tile_x, u32 tile_y,
                                            u32 row_stride) {
  const unsigned long long ga = (unsigned long long)(uintptr_t)gptr;
  const u32 lds = (u32)(uintptr_t)lptr;
  v4u g0;
  g0[0] = 1u;                                          // count=1, user mode
  g0[1] = lds;                                         // lds_addr [63:32]
  g0[2] = (u32)ga;                                     // global_addr[31:0]
  g0[3] = (u32)((ga >> 32) & 0x01FFFFFFu) | (2u << 30);// addr[56:32] | type=2
  const u32 td0 = row_stride;                          // tensor_dim0 >= tile_x
  const u32 td1 = 0x00100000u;                         // tensor_dim1: large
  v8i g1;
  // data_size=1 (2B) | pad_enable | pad_interval=4 (32dw) | pad_amount=3 (4dw)
  g1[0] = (int)((1u << 16) | (1u << 20) | (4u << 22) | (3u << 25));
  g1[1] = (int)((td0 & 0xFFFFu) << 16);                // tensor_dim0 lo
  g1[2] = (int)((td0 >> 16) | ((td1 & 0xFFFFu) << 16));// dim0 hi | dim1 lo
  g1[3] = (int)((td1 >> 16) | (tile_x << 16));         // dim1 hi | tile_dim0
  g1[4] = (int)(tile_y & 0xFFFFu);                     // tile_dim1 (tile_dim2=0)
  g1[5] = (int)row_stride;                             // tensor_dim0_stride lo
  g1[6] = 0;                                           // stride hi | dim1_str lo
  g1[7] = 0;
  const v4i z4 = {0, 0, 0, 0};
  const v8i z8 = {0, 0, 0, 0, 0, 0, 0, 0};
  __builtin_amdgcn_tensor_load_to_lds(g0, g1, z4, z4, z8, 0);
}
__device__ __forceinline__ void wait_tensor() {
  __builtin_amdgcn_s_wait_tensorcnt(0);
}

// ---- DPP 16-lane butterfly reductions (stay within one DPP row == half-wave)
template <int CTRL>
__device__ __forceinline__ float dpp_perm(float x) {
  return __builtin_bit_cast(
      float, __builtin_amdgcn_update_dpp(0, __builtin_bit_cast(int, x), CTRL,
                                         0xF, 0xF, true));
}
__device__ __forceinline__ float red16_max(float x) {
  x = fmaxf(x, dpp_perm<0xB1>(x));    // quad_perm(1,0,3,2)  : xor1
  x = fmaxf(x, dpp_perm<0x4E>(x));    // quad_perm(2,3,0,1)  : xor2
  x = fmaxf(x, dpp_perm<0x141>(x));   // row_half_mirror     : mix quads in 8
  x = fmaxf(x, dpp_perm<0x140>(x));   // row_mirror          : mix 8s in 16
  return x;
}
__device__ __forceinline__ float red16_sum(float x) {
  x += dpp_perm<0xB1>(x);
  x += dpp_perm<0x4E>(x);
  x += dpp_perm<0x141>(x);
  x += dpp_perm<0x140>(x);
  return x;
}

// ---------------- f32 -> bf16 convert ----------------
__global__ __launch_bounds__(256) void cvt_kernel(const float* __restrict__ in,
                                                  bf16* __restrict__ out, int n) {
  int i = blockIdx.x * 256 + threadIdx.x;
  if (i < n) out[i] = (bf16)in[i];
}

// ---------------- fused LayerNorm (one row per block) ----------------
template <bool WRITE_RAW>
__global__ __launch_bounds__(256) void ln_kernel(const float* __restrict__ x,
                                                 const float* __restrict__ g,
                                                 const float* __restrict__ bb,
                                                 bf16* __restrict__ lnout,
                                                 bf16* __restrict__ rawout) {
  const int row = blockIdx.x, tid = threadIdx.x;
  const float* xr = x + (size_t)row * kH;
  float v[3];
#pragma unroll
  for (int i = 0; i < 3; i++) v[i] = xr[tid + i * 256];
  float s = v[0] + v[1] + v[2];
  float q = v[0] * v[0] + v[1] * v[1] + v[2] * v[2];
#pragma unroll
  for (int off = 16; off >= 1; off >>= 1) {
    s += __shfl_xor(s, off, 32);
    q += __shfl_xor(q, off, 32);
  }
  __shared__ float red[16];
  const int wave = tid >> 5, lane = tid & 31;
  if (lane == 0) { red[wave] = s; red[8 + wave] = q; }
  __syncthreads();
  float S_ = 0.f, Q_ = 0.f;
#pragma unroll
  for (int w = 0; w < 8; w++) { S_ += red[w]; Q_ += red[8 + w]; }
  const float mean = S_ * (1.0f / (float)kH);
  const float var  = Q_ * (1.0f / (float)kH) - mean * mean;
  const float rstd = rsqrtf(var + 1e-12f);
#pragma unroll
  for (int i = 0; i < 3; i++) {
    const int c = tid + i * 256;
    lnout[(size_t)row * kH + c] = (bf16)((v[i] - mean) * rstd * g[c] + bb[c]);
    if (WRITE_RAW) rawout[(size_t)row * kH + c] = (bf16)v[i];
  }
}

// ---------------- bf16 WMMA GEMM: C = (A@W + bias) [*oscale | *lam + res] ---
// block = 256 thr (8 waves), tile 128x128, BK=64; wave tile 32x64 (2x4 frags).
// A tile staged by TDM (wave 0), W tile transposed manually into [n][k].
template <bool HAS_BIAS, bool SCALE_RES>
__global__ __launch_bounds__(256) void gemm_bf16_kernel(
    const bf16* __restrict__ A, const bf16* __restrict__ W,
    const float* __restrict__ bias, const float* __restrict__ scale,
    const float* __restrict__ res, void* __restrict__ Cout,
    int M, int N, int K, float oscale) {
  constexpr int BM = 128, BN = 128, BK = 64, LS = 72;  // LS: padded bf16 stride
  __shared__ __align__(16) bf16 Alds[BM * LS];  // row-major [m][k] (TDM-padded)
  __shared__ __align__(16) bf16 Blds[BN * LS];  // column-major [n][k]
  const int tid = threadIdx.x;
  const int wave = tid >> 5, lane = tid & 31;
  const int bm = blockIdx.x * BM, bn = blockIdx.y * BN;
  const int wm = (wave & 3) * 32, wn = (wave >> 2) * 64;

  v8f acc[2][4];
#pragma unroll
  for (int i = 0; i < 2; i++)
#pragma unroll
    for (int j = 0; j < 4; j++)
#pragma unroll
      for (int r = 0; r < 8; r++) acc[i][j][r] = 0.f;

  for (int k0 = 0; k0 < K; k0 += BK) {
    // A tile 128x64 via Tensor Data Mover (one descriptor, wave 0 only)
    if (wave == 0)
      tdm_load_2d(A + (size_t)bm * K + k0, Alds, BK, BM, (u32)K);
    // W tile 64x128: coalesced row-pair loads, transpose into [n][k] as b32
#pragma unroll
    for (int p = 0; p < 2; p++) {
      const int linear = p * 256 + tid;
      const int rp = linear >> 4;          // k-pair 0..31 -> k = 2*rp
      const int c  = (linear & 15) * 8;    // n start
      const bf16* g0 = W + (size_t)(k0 + 2 * rp) * N + (bn + c);
      uint4 d0 = *(const uint4*)g0;
      uint4 d1 = *(const uint4*)(g0 + N);
      const bf16* a0 = reinterpret_cast<const bf16*>(&d0);
      const bf16* a1 = reinterpret_cast<const bf16*>(&d1);
#pragma unroll
      for (int j = 0; j < 8; j++) {
        union { bf16 b[2]; uint32_t u; } t;
        t.b[0] = a0[j];
        t.b[1] = a1[j];
        *(uint32_t*)(&Blds[(c + j) * LS + 2 * rp]) = t.u;
      }
    }
    if (wave == 0) wait_tensor();
    __syncthreads();
#pragma unroll
    for (int kk = 0; kk < 2; kk++) {
      v16bf a[2], b[4];
#pragma unroll
      for (int i = 0; i < 2; i++)
        a[i] = frag_a_lds(&Alds[(wm + i * 16) * LS + kk * 32], lane, LS);
#pragma unroll
      for (int j = 0; j < 4; j++)
        b[j] = frag_b_lds(&Blds[(wn + j * 16) * LS + kk * 32], lane, LS);
#pragma unroll
      for (int i = 0; i < 2; i++)
#pragma unroll
        for (int j = 0; j < 4; j++) acc[i][j] = wmma_bf16(a[i], b[j], acc[i][j]);
    }
    __syncthreads();
  }
  // epilogue (C/D layout: lanes 0-15 rows 0..7, lanes 16-31 rows 8..15)
  const int nl = lane & 15, hi8 = lane >> 4;
#pragma unroll
  for (int i = 0; i < 2; i++)
#pragma unroll
    for (int j = 0; j < 4; j++)
#pragma unroll
      for (int r = 0; r < 8; r++) {
        const int row = bm + wm + i * 16 + hi8 * 8 + r;
        const int col = bn + wn + j * 16 + nl;
        float v = acc[i][j][r];
        if (HAS_BIAS) v += bias[col];
        if (SCALE_RES) {
          ((float*)Cout)[(size_t)row * N + col] =
              v * scale[col] + res[(size_t)row * N + col];
        } else {
          ((bf16*)Cout)[(size_t)row * N + col] = (bf16)(v * oscale);
        }
      }
}

// ---------------- flash attention (per (b,head), 128 q rows / block) -------
// q is pre-scaled by 1/sqrt(HD) in the Q-projection epilogue.
// Q tile staged by TDM; K tile by per-lane async-to-LDS; V transposed manually.
__global__ __launch_bounds__(256) void attn_kernel(const bf16* __restrict__ q,
                                                   const bf16* __restrict__ k,
                                                   const bf16* __restrict__ v,
                                                   bf16* __restrict__ ctx) {
  constexpr int LS = 72;
  __shared__ __align__(16) bf16 Qs[128 * LS];      // [q][hd] (TDM-padded)
  __shared__ __align__(16) bf16 Ks[64 * LS];       // [kv][hd] == B col-major
  __shared__ __align__(16) bf16 Vt[64 * LS];       // [hd][kv] == B col-major
  __shared__ __align__(16) bf16 Ps[8 * 16 * LS];   // per-wave P [16][kv]
  const int tid = threadIdx.x, wave = tid >> 5, lane = tid & 31;
  const int bz = blockIdx.x / kNH, hh = blockIdx.x % kNH;
  const int q0 = blockIdx.y * 128;
  const bf16* qbase = q + ((size_t)bz * kS) * kH + (size_t)hh * kHD;
  const bf16* kbase = k + ((size_t)bz * kS) * kH + (size_t)hh * kHD;
  const bf16* vbase = v + ((size_t)bz * kS) * kH + (size_t)hh * kHD;
  const int nl = lane & 15, hi8 = lane >> 4;

  // Q tile 128x64 via TDM (wave 0)
  if (wave == 0)
    tdm_load_2d(qbase + (size_t)q0 * kH, Qs, kHD, 128, (u32)kH);

  const int wq = wave * 16;
  v8f O[4];
  float mi[8], li[8];
#pragma unroll
  for (int j = 0; j < 4; j++)
#pragma unroll
    for (int r = 0; r < 8; r++) O[j][r] = 0.f;
#pragma unroll
  for (int r = 0; r < 8; r++) { mi[r] = -1e30f; li[r] = 0.f; }

  for (int kt = 0; kt < kS; kt += 64) {
    // K tile 64x64: per-lane async DMA row-major
#pragma unroll
    for (int p = 0; p < 2; p++) {
      const int linear = p * 256 + tid;
      const int r = linear >> 3, c = (linear & 7) * 8;
      async_copy_b128(kbase + (size_t)(kt + r) * kH + c, &Ks[r * LS + c]);
    }
    // V tile 64x64: row-pair loads, transpose into [hd][kv] as b32 stores
    {
      const int rp = tid >> 3;           // kv-pair 0..31 -> kv = 2*rp
      const int c  = (tid & 7) * 8;      // hd start
      const bf16* g0 = vbase + (size_t)(kt + 2 * rp) * kH + c;
      uint4 d0 = *(const uint4*)g0;
      uint4 d1 = *(const uint4*)(g0 + kH);
      const bf16* a0 = reinterpret_cast<const bf16*>(&d0);
      const bf16* a1 = reinterpret_cast<const bf16*>(&d1);
#pragma unroll
      for (int j = 0; j < 8; j++) {
        union { bf16 b[2]; uint32_t u; } t;
        t.b[0] = a0[j];
        t.b[1] = a1[j];
        *(uint32_t*)(&Vt[(c + j) * LS + 2 * rp]) = t.u;
      }
    }
    wait_async();
    if (wave == 0) wait_tensor();
    __syncthreads();

    // scores: S = Q @ K^T   (16q x 64kv per wave), Q pre-scaled
    v8f sc[4];
#pragma unroll
    for (int j = 0; j < 4; j++)
#pragma unroll
      for (int r = 0; r < 8; r++) sc[j][r] = 0.f;
#pragma unroll
    for (int kk = 0; kk < 2; kk++) {
      v16bf a = frag_a_lds(&Qs[wq * LS + kk * 32], lane, LS);
#pragma unroll
      for (int j = 0; j < 4; j++) {
        v16bf bfr = frag_b_lds(&Ks[(j * 16) * LS + kk * 32], lane, LS);
        sc[j] = wmma_bf16(a, bfr, sc[j]);
      }
    }

    // online softmax: rows live in half-wave lane groups -> DPP reductions
    float alpha[8];
#pragma unroll
    for (int r = 0; r < 8; r++) {
      float mx = fmaxf(fmaxf(sc[0][r], sc[1][r]), fmaxf(sc[2][r], sc[3][r]));
      mx = red16_max(mx);
      const float mnew = fmaxf(mi[r], mx);
      alpha[r] = __expf(mi[r] - mnew);
      mi[r] = mnew;
    }
#pragma unroll
    for (int r = 0; r < 8; r++) {
      float rs = 0.f;
#pragma unroll
      for (int j = 0; j < 4; j++) {
        const float pv = __expf(sc[j][r] - mi[r]);
        sc[j][r] = pv;
        rs += pv;
      }
      rs = red16_sum(rs);
      li[r] = li[r] * alpha[r] + rs;
#pragma unroll
      for (int j = 0; j < 4; j++) O[j][r] *= alpha[r];
    }

    // C-layout -> A-layout via per-wave LDS region (same-wave DS is in-order)
    bf16* pw = &Ps[wave * 16 * LS];
#pragma unroll
    for (int j = 0; j < 4; j++)
#pragma unroll
      for (int r = 0; r < 8; r++)
        pw[(hi8 * 8 + r) * LS + j * 16 + nl] = (bf16)sc[j][r];

    // O += P @ V
#pragma unroll
    for (int kk = 0; kk < 2; kk++) {
      v16bf pa = frag_a_lds(pw + kk * 32, lane, LS);
#pragma unroll
      for (int j = 0; j < 4; j++) {
        v16bf bv = frag_b_lds(&Vt[(j * 16) * LS + kk * 32], lane, LS);
        O[j] = wmma_bf16(pa, bv, O[j]);
      }
    }
    __syncthreads();
  }

  // normalize and emit ctx (bf16, [B,S,NH,HD] flattened as [M, H])
#pragma unroll
  for (int j = 0; j < 4; j++)
#pragma unroll
    for (int r = 0; r < 8; r++) {
      const int row = q0 + wq + hi8 * 8 + r;
      const int col = hh * kHD + j * 16 + nl;
      ctx[((size_t)bz * kS + row) * kH + col] = (bf16)(O[j][r] / li[r]);
    }
}

// ---------------- launcher ----------------
extern "C" void kernel_launch(void* const* d_in, const int* in_sizes, int n_in,
                              void* d_out, int out_size, void* d_ws, size_t ws_size,
                              hipStream_t stream) {
  const float* x    = (const float*)d_in[0];
  const float* wq   = (const float*)d_in[1];
  const float* bq   = (const float*)d_in[2];
  const float* wk   = (const float*)d_in[3];
  const float* wv   = (const float*)d_in[4];
  const float* bv   = (const float*)d_in[5];
  const float* wo   = (const float*)d_in[6];
  const float* bo   = (const float*)d_in[7];
  const float* wi   = (const float*)d_in[8];
  const float* bi   = (const float*)d_in[9];
  const float* w2   = (const float*)d_in[10];
  const float* b2   = (const float*)d_in[11];
  const float* ln1g = (const float*)d_in[12];
  const float* ln1b = (const float*)d_in[13];
  const float* ln2g = (const float*)d_in[14];
  const float* ln2b = (const float*)d_in[15];
  const float* lam1 = (const float*)d_in[16];
  const float* lam2 = (const float*)d_in[17];

  char* p = (char*)d_ws;
  auto alloc = [&](size_t bytes) -> char* {
    char* r = p;
    p += (bytes + 255) & ~(size_t)255;
    return r;
  };
  const size_t actB = (size_t)kM * kH * sizeof(bf16);  // 6.29 MB, 256-aligned
  bf16*  xb   = (bf16*)alloc(actB);
  bf16*  lnx  = (bf16*)alloc(actB);     // reused for LN2 output
  bf16*  qb   = (bf16*)alloc(actB);     // q|k|v|ctx region reused as FF interm
  bf16*  kb   = (bf16*)alloc(actB);
  bf16*  vb   = (bf16*)alloc(actB);
  bf16*  ctxb = (bf16*)alloc(actB);
  float* hb   = (float*)alloc((size_t)kM * kH * sizeof(float));
  bf16*  wqb  = (bf16*)alloc((size_t)kH * kH * sizeof(bf16));
  bf16*  wkb  = (bf16*)alloc((size_t)kH * kH * sizeof(bf16));
  bf16*  wvb  = (bf16*)alloc((size_t)kH * kH * sizeof(bf16));
  bf16*  wob  = (bf16*)alloc((size_t)kH * kH * sizeof(bf16));
  bf16*  wib  = (bf16*)alloc((size_t)kH * kFF * sizeof(bf16));
  bf16*  w2b  = (bf16*)alloc((size_t)kFF * kH * sizeof(bf16));
  bf16*  ffib = qb;  // 4 * actB == kM * kFF * 2 bytes exactly

  const int nHH = kH * kH, nHF = kH * kFF;
  cvt_kernel<<<(nHH + 255) / 256, 256, 0, stream>>>(wq, wqb, nHH);
  cvt_kernel<<<(nHH + 255) / 256, 256, 0, stream>>>(wk, wkb, nHH);
  cvt_kernel<<<(nHH + 255) / 256, 256, 0, stream>>>(wv, wvb, nHH);
  cvt_kernel<<<(nHH + 255) / 256, 256, 0, stream>>>(wo, wob, nHH);
  cvt_kernel<<<(nHF + 255) / 256, 256, 0, stream>>>(wi, wib, nHF);
  cvt_kernel<<<(nHF + 255) / 256, 256, 0, stream>>>(w2, w2b, nHF);

  // LN1 + raw x in bf16
  ln_kernel<true><<<kM, 256, 0, stream>>>(x, ln1g, ln1b, lnx, xb);

  // Q/K/V projections (Q pre-scaled by 1/sqrt(HD))
  gemm_bf16_kernel<true,  false><<<dim3(kM / 128, kH / 128), 256, 0, stream>>>(
      lnx, wqb, bq, nullptr, nullptr, qb, kM, kH, kH, 0.125f);
  gemm_bf16_kernel<false, false><<<dim3(kM / 128, kH / 128), 256, 0, stream>>>(
      xb, wkb, nullptr, nullptr, nullptr, kb, kM, kH, kH, 1.0f);
  gemm_bf16_kernel<true,  false><<<dim3(kM / 128, kH / 128), 256, 0, stream>>>(
      xb, wvb, bv, nullptr, nullptr, vb, kM, kH, kH, 1.0f);

  // attention
  attn_kernel<<<dim3(kB * kNH, kS / 128), 256, 0, stream>>>(qb, kb, vb, ctxb);

  // h = (ctx @ wo + bo) * lambda1 + x      (f32)
  gemm_bf16_kernel<true, true><<<dim3(kM / 128, kH / 128), 256, 0, stream>>>(
      ctxb, wob, bo, lam1, x, hb, kM, kH, kH, 1.0f);

  // LN2
  ln_kernel<false><<<kM, 256, 0, stream>>>(hb, ln2g, ln2b, lnx, nullptr);

  // FF intermediate (bf16)
  gemm_bf16_kernel<true, false><<<dim3(kM / 128, kFF / 128), 256, 0, stream>>>(
      lnx, wib, bi, nullptr, nullptr, ffib, kM, kFF, kH, 1.0f);

  // out = (ffi @ w2 + b2) * lambda2 + h    (f32 -> d_out)
  gemm_bf16_kernel<true, true><<<dim3(kM / 128, kH / 128), 256, 0, stream>>>(
      ffib, w2b, b2, lam2, hb, (float*)d_out, kM, kH, kFF, 1.0f);
}
